// NeighborhoodAttention_85495618994612
// MI455X (gfx1250) — matrix operand order repaired and checked by reading, not compiled, not run on hardware
//
#include <hip/hip_runtime.h>
#include <hip/hip_bf16.h>
#include <math.h>

#define HEADS 8
#define HD    54
#define CH    432
#define HWD   56
#define NPIX  3136
#define KW    7
#define KK    49
#define QKVN  1296
#define C2O   392

#define BM 64
#define BN 128
#define BK 32

typedef __attribute__((ext_vector_type(16))) __bf16        v16bf;
typedef __attribute__((ext_vector_type(8)))  float         v8f;
typedef __attribute__((ext_vector_type(8)))  unsigned int  v8u;

// ---------------------------------------------------------------------------
// Pipelined LDS-staged bf16 WMMA GEMM:  C[M,N] = A[M,K] * B + bias
//   B_NK==1 : B indexed as B[n*ldb + k]   (weights stored (N,K))
//   B_NK==0 : B indexed as B[k*ldb + n]   (activations stored (K,N))
//   BIAS_MODE: 0 none, 1 per-column (N), 2 per-row (M)
// Block (256 thr / 8 waves) computes a 64x128 C tile; each wave owns 4
// 16x16 subtiles sharing one A fragment (4 v_wmma per chunk per wave).
// Software pipeline: global loads for chunk t+1 issue before the WMMAs of
// chunk t, so HBM latency hides behind the matrix pipe.
// ---------------------------------------------------------------------------
template<int B_NK, int BIAS_MODE>
__global__ __launch_bounds__(256) void gemm_bf16_wmma(
    const float* __restrict__ A, int lda,
    const float* __restrict__ B, int ldb,
    const float* __restrict__ bias,
    float* __restrict__ C, int ldc,
    int M, int N, int K)
{
  __shared__ __bf16 As[BM * BK];      // As[m][k]
  __shared__ __bf16 Bs[BN * BK];      // Bs[n][k]

  const int tid  = threadIdx.x;
  const int wave = tid >> 5;
  const int lane = tid & 31;
  const int half = lane >> 4;
  const int l15  = lane & 15;

  const int bm = blockIdx.x * BM;
  const int bn = blockIdx.y * BN;
  const bool fullM = (bm + BM <= M);
  const bool fullN = (bn + BN <= N);

  // wave -> subtiles: wm in 0..3 (M); n-subtiles wnb, wnb+2, wnb+4, wnb+6
  const int wm  = wave & 3;
  const int wnb = wave >> 2;
  v8f acc0 = {}, acc1 = {}, acc2 = {}, acc3 = {};

  // A staging: 64 rows x 4 thr/row x 8 elems
  const int arow = tid >> 2;
  const int akc  = (tid & 3) << 3;
  // B_NK staging: 128 rows x 2 thr/row x 16 elems
  const int brow = tid >> 1;
  const int bkc  = (tid & 1) << 4;
  // B_KN staging: 32 k-rows x 8 thr/row x 16 n-elems
  const int bkr  = tid >> 3;
  const int bnc  = (tid & 7) << 4;

  float ra[8];
  float rb[16];

  auto loadA = [&](int kb) {
    const int rg = bm + arow;
    if (fullM && kb + BK <= K) {
      const float4 f0 = *reinterpret_cast<const float4*>(A + (size_t)rg * lda + kb + akc);
      const float4 f1 = *reinterpret_cast<const float4*>(A + (size_t)rg * lda + kb + akc + 4);
      ra[0]=f0.x; ra[1]=f0.y; ra[2]=f0.z; ra[3]=f0.w;
      ra[4]=f1.x; ra[5]=f1.y; ra[6]=f1.z; ra[7]=f1.w;
    } else {
      #pragma unroll
      for (int q = 0; q < 8; ++q) {
        const int kg = kb + akc + q;
        ra[q] = (rg < M && kg < K) ? A[(size_t)rg * lda + kg] : 0.f;
      }
    }
  };

  auto loadB = [&](int kb) {
    if (B_NK) {
      const int ng = bn + brow;
      if (fullN && kb + BK <= K) {
        #pragma unroll
        for (int h = 0; h < 4; ++h) {
          const float4 f = *reinterpret_cast<const float4*>(B + (size_t)ng * ldb + kb + bkc + 4 * h);
          rb[4*h+0]=f.x; rb[4*h+1]=f.y; rb[4*h+2]=f.z; rb[4*h+3]=f.w;
        }
      } else {
        #pragma unroll
        for (int q = 0; q < 16; ++q) {
          const int kg = kb + bkc + q;
          rb[q] = (ng < N && kg < K) ? B[(size_t)ng * ldb + kg] : 0.f;
        }
      }
    } else {
      const int kg = kb + bkr;
      if (fullN && kb + BK <= K) {
        #pragma unroll
        for (int h = 0; h < 4; ++h) {
          const float4 f = *reinterpret_cast<const float4*>(B + (size_t)kg * ldb + bn + bnc + 4 * h);
          rb[4*h+0]=f.x; rb[4*h+1]=f.y; rb[4*h+2]=f.z; rb[4*h+3]=f.w;
        }
      } else {
        #pragma unroll
        for (int q = 0; q < 16; ++q) {
          const int ng = bn + bnc + q;
          rb[q] = (kg < K && ng < N) ? B[(size_t)kg * ldb + ng] : 0.f;
        }
      }
    }
  };

  // ---- pipeline prologue
  loadA(0);
  loadB(0);

  for (int kb = 0; kb < K; kb += BK) {
    __syncthreads();              // previous chunk's LDS reads complete
    // ---- commit staged regs to LDS (waits on the in-flight global loads)
    {
      __bf16* da = &As[arow * BK + akc];
      #pragma unroll
      for (int q = 0; q < 8; ++q) da[q] = (__bf16)ra[q];
      if (B_NK) {
        __bf16* db = &Bs[brow * BK + bkc];
        #pragma unroll
        for (int q = 0; q < 16; ++q) db[q] = (__bf16)rb[q];
      } else {
        #pragma unroll
        for (int q = 0; q < 16; ++q) Bs[(bnc + q) * BK + bkr] = (__bf16)rb[q];
      }
    }
    __syncthreads();

    // ---- issue next chunk's global loads (overlap with WMMA below)
    if (kb + BK < K) { loadA(kb + BK); loadB(kb + BK); }

    // ---- A fragment (ISA 7.12.2 16-bit A swizzle)
    v8u au;
    #pragma unroll
    for (int v = 0; v < 8; ++v) {
      const int ka = ((v >= 4) ? 16 : 0) + (half << 3) + ((v & 3) << 1);
      au[v] = *reinterpret_cast<const unsigned int*>(&As[(wm * 16 + l15) * BK + ka]);
    }
    const v16bf a = __builtin_bit_cast(v16bf, au);

    // ---- 4 B fragments + 4 WMMAs
    v8u bu0, bu1, bu2, bu3;
    #pragma unroll
    for (int v = 0; v < 8; ++v) {
      const int kf = ((v >= 4) ? 16 : 0) + (half << 3) + ((v & 3) << 1);
      bu0[v] = *reinterpret_cast<const unsigned int*>(&Bs[((wnb + 0) * 16 + l15) * BK + kf]);
      bu1[v] = *reinterpret_cast<const unsigned int*>(&Bs[((wnb + 2) * 16 + l15) * BK + kf]);
      bu2[v] = *reinterpret_cast<const unsigned int*>(&Bs[((wnb + 4) * 16 + l15) * BK + kf]);
      bu3[v] = *reinterpret_cast<const unsigned int*>(&Bs[((wnb + 6) * 16 + l15) * BK + kf]);
    }
    acc0 = __builtin_amdgcn_wmma_f32_16x16x32_bf16(false, a, false, __builtin_bit_cast(v16bf, bu0), (short)0, acc0, false, false);
    acc1 = __builtin_amdgcn_wmma_f32_16x16x32_bf16(false, a, false, __builtin_bit_cast(v16bf, bu1), (short)0, acc1, false, false);
    acc2 = __builtin_amdgcn_wmma_f32_16x16x32_bf16(false, a, false, __builtin_bit_cast(v16bf, bu2), (short)0, acc2, false, false);
    acc3 = __builtin_amdgcn_wmma_f32_16x16x32_bf16(false, a, false, __builtin_bit_cast(v16bf, bu3), (short)0, acc3, false, false);
  }

  // ---- writeback: C/D layout VGPR r -> M = half*8+r, N = lane&15
  #pragma unroll
  for (int r = 0; r < 8; ++r) {
    const int row = bm + wm * 16 + (half << 3) + r;
    if (row >= M) continue;
    const float brr = (BIAS_MODE == 2) ? bias[row] : 0.f;
    #pragma unroll
    for (int s = 0; s < 4; ++s) {
      const int col = bn + (wnb + 2 * s) * 16 + l15;
      if (col >= N) continue;
      float v = (s == 0 ? acc0[r] : s == 1 ? acc1[r] : s == 2 ? acc2[r] : acc3[r]);
      if (BIAS_MODE == 1) v += bias[col];
      if (BIAS_MODE == 2) v += brr;
      C[(size_t)row * ldc + col] = v;
    }
  }
}

// ---------------------------------------------------------------------------
// Neighborhood attention logits + rpb + softmax.  One thread per (head,pixel).
// qkv layout: [pixel][1296] with q at +0, k at +432, v at +864. q unscaled.
// ---------------------------------------------------------------------------
__global__ __launch_bounds__(256) void attn_softmax_kernel(
    const float* __restrict__ qkv, const float* __restrict__ rpb,
    float* __restrict__ attn)
{
  const int t = blockIdx.x * blockDim.x + threadIdx.x;
  if (t >= HEADS * NPIX) return;
  const int g = t / NPIX, p = t % NPIX;
  const int i = p / HWD, j = p % HWD;
  const int ni = min(max(i - 3, 0), HWD - KW);
  const int nj = min(max(j - 3, 0), HWD - KW);
  const float scale = 0.13608276348795434f;   // 54^-0.5

  const float* qp = qkv + (size_t)p * QKVN + g * HD;
  float logits[KK];
  float mx = -1e30f;
  for (int u = 0; u < KW; ++u) {
    const int ri = 6 - (i - ni) + u;
    for (int w = 0; w < KW; ++w) {
      const int rj = 6 - (j - nj) + w;
      const int pn = (ni + u) * HWD + (nj + w);
      const float* kp = qkv + (size_t)pn * QKVN + CH + g * HD;
      float s = 0.f;
      for (int d = 0; d < HD; ++d) s += qp[d] * kp[d];
      s = s * scale + rpb[g * 169 + ri * 13 + rj];
      logits[u * KW + w] = s;
      mx = fmaxf(mx, s);
    }
  }
  float sum = 0.f;
  for (int kk = 0; kk < KK; ++kk) { float e = __expf(logits[kk] - mx); logits[kk] = e; sum += e; }
  const float inv = 1.f / sum;
  float* ap = attn + (size_t)t * KK;
  for (int kk = 0; kk < KK; ++kk) ap[kk] = logits[kk] * inv;
}

// had[c][p] = q_raw * k / 54
__global__ __launch_bounds__(256) void had_kernel(
    const float* __restrict__ qkv, float* __restrict__ had)
{
  const int t = blockIdx.x * blockDim.x + threadIdx.x;
  if (t >= CH * NPIX) return;
  const int c = t / NPIX, p = t % NPIX;
  const float q = qkv[(size_t)p * QKVN + c];
  const float k = qkv[(size_t)p * QKVN + CH + c];
  had[t] = q * k * 0.018518518518518517f;
}

// Grouped 7x7 conv (54 groups of 8 in/out channels), zero pad 3, + bias, GELU.
__global__ __launch_bounds__(256) void conv1_gelu_kernel(
    const float* __restrict__ had, const float* __restrict__ w,
    const float* __restrict__ b, float* __restrict__ h1)
{
  const int t = blockIdx.x * blockDim.x + threadIdx.x;
  if (t >= CH * NPIX) return;
  const int o = t / NPIX, p = t % NPIX;
  const int i = p / HWD, j = p % HWD;
  const int cbase = (o >> 3) << 3;
  float s = b[o];
  for (int ci = 0; ci < 8; ++ci) {
    const float* hp = had + (size_t)(cbase + ci) * NPIX;
    const float* wp = w + (size_t)o * (8 * KK) + ci * KK;
    for (int u = 0; u < KW; ++u) {
      const int y = i - 3 + u;
      if (y < 0 || y >= HWD) continue;
      for (int wv = 0; wv < KW; ++wv) {
        const int x = j - 3 + wv;
        if (x < 0 || x >= HWD) continue;
        s += wp[u * KW + wv] * hp[y * HWD + x];
      }
    }
  }
  h1[t] = 0.5f * s * (1.f + erff(s * 0.70710678118654752f));
}

// Softmax over 392 channels per pixel; scatter to h_attn[(g,p,kk)] layout.
__global__ __launch_bounds__(256) void chsoftmax_kernel(
    const float* __restrict__ h2, float* __restrict__ hattn)
{
  const int p = blockIdx.x * blockDim.x + threadIdx.x;
  if (p >= NPIX) return;
  float mx = -1e30f;
  for (int c = 0; c < C2O; ++c) mx = fmaxf(mx, h2[(size_t)c * NPIX + p]);
  float sum = 0.f;
  for (int c = 0; c < C2O; ++c) {
    const float e = __expf(h2[(size_t)c * NPIX + p] - mx);
    sum += e;
    hattn[(((size_t)(c / KK)) * NPIX + p) * KK + (c % KK)] = e;
  }
  const float inv = 1.f / sum;
  for (int c = 0; c < C2O; ++c)
    hattn[(((size_t)(c / KK)) * NPIX + p) * KK + (c % KK)] *= inv;
}

// Dual aggregation: av (clipped gather, attn+h_attn) and elsa (zero-pad
// gather, attn for d<27 else h_attn + ghost). out2 (3136, 864).
__global__ __launch_bounds__(256) void aggregate_kernel(
    const float* __restrict__ qkv, const float* __restrict__ attn,
    const float* __restrict__ hattn, const float* __restrict__ ghost,
    float* __restrict__ out2)
{
  const int t = blockIdx.x * blockDim.x + threadIdx.x;
  if (t >= CH * NPIX) return;
  const int c = t / NPIX, p = t % NPIX;
  const int g = c / HD, d = c % HD;
  const int i = p / HWD, j = p % HWD;
  const int ni = min(max(i - 3, 0), HWD - KW);
  const int nj = min(max(j - 3, 0), HWD - KW);

  const float* at = attn  + ((size_t)g * NPIX + p) * KK;
  const float* ht = hattn + ((size_t)g * NPIX + p) * KK;
  const float* gh = ghost + (size_t)c * KK;
  const float* ga = (d < 27) ? at : ht;

  float av = 0.f, el = 0.f;
  for (int u = 0; u < KW; ++u) {
    const int yc = ni + u;
    const int yz = i - 3 + u;
    for (int w = 0; w < KW; ++w) {
      const int kk = u * KW + w;
      const int pc = yc * HWD + (nj + w);
      av += (at[kk] + ht[kk]) * qkv[(size_t)pc * QKVN + 2 * CH + c];
      const int xz = j - 3 + w;
      float vv = 0.f;
      if (yz >= 0 && yz < HWD && xz >= 0 && xz < HWD)
        vv = qkv[(size_t)(yz * HWD + xz) * QKVN + 2 * CH + c];
      el += vv * (ga[kk] + gh[kk]);
    }
  }
  float* op = out2 + (size_t)p * (2 * CH) + g * (2 * HD);
  op[d]      = av;
  op[HD + d] = el;
}

// ---------------------------------------------------------------------------
extern "C" void kernel_launch(void* const* d_in, const int* in_sizes, int n_in,
                              void* d_out, int out_size, void* d_ws, size_t ws_size,
                              hipStream_t stream)
{
  const float* x       = (const float*)d_in[0];
  const float* qkv_w   = (const float*)d_in[1];   // (1296, 432)
  const float* qkv_b   = (const float*)d_in[2];
  const float* rpb     = (const float*)d_in[3];   // (8, 13, 13)
  const float* conv1_w = (const float*)d_in[4];   // (432, 8, 7, 7)
  const float* conv1_b = (const float*)d_in[5];
  const float* conv2_w = (const float*)d_in[6];   // (392, 432)
  const float* conv2_b = (const float*)d_in[7];
  const float* ghost   = (const float*)d_in[8];   // (432, 7, 7)
  const float* proj2_w = (const float*)d_in[9];   // (432, 864)
  const float* proj2_b = (const float*)d_in[10];
  float* out = (float*)d_out;                     // (3136, 432) f32

  float* ws    = (float*)d_ws;
  float* qkv   = ws;                       // 3136*1296 = 4,064,256
  float* attn  = qkv   + 4064256;          // 8*3136*49 = 1,229,312
  float* had   = attn  + 1229312;          // 432*3136  = 1,354,752
  float* h1    = had   + 1354752;
  float* h2    = h1    + 1354752;          // 392*3136  = 1,229,312
  float* hattn = h2    + 1229312;
  float* out2  = hattn + 1229312;          // 3136*864  = 2,709,504

  auto gdim = [](int M, int N) {
    return dim3((M + BM - 1) / BM, (N + BN - 1) / BN, 1);
  };

  // 1) QKV projection: (3136,432)x(432,1296)+b -> qkv
  gemm_bf16_wmma<1, 1><<<gdim(NPIX, QKVN), 256, 0, stream>>>(
      x, CH, qkv_w, CH, qkv_b, qkv, QKVN, NPIX, QKVN, CH);

  // 2) neighborhood attention softmax
  attn_softmax_kernel<<<(HEADS * NPIX + 255) / 256, 256, 0, stream>>>(qkv, rpb, attn);

  // 3) had = q*k/54 (NCHW)
  had_kernel<<<(CH * NPIX + 255) / 256, 256, 0, stream>>>(qkv, had);

  // 4) grouped conv 7x7 + GELU
  conv1_gelu_kernel<<<(CH * NPIX + 255) / 256, 256, 0, stream>>>(had, conv1_w, conv1_b, h1);

  // 5) 1x1 conv as GEMM: (392,432)x(432,3136)+b_row -> h2
  gemm_bf16_wmma<0, 2><<<gdim(C2O, NPIX), 256, 0, stream>>>(
      conv2_w, CH, h1, NPIX, conv2_b, h2, NPIX, C2O, NPIX, CH);

  // 6) channel softmax -> h_attn
  chsoftmax_kernel<<<(NPIX + 255) / 256, 256, 0, stream>>>(h2, hattn);

  // 7) av + elsa(+ghost) aggregation -> out2 (3136, 864)
  aggregate_kernel<<<(CH * NPIX + 255) / 256, 256, 0, stream>>>(qkv, attn, hattn, ghost, out2);

  // 8) final projection: (3136,864)x(864,432)+b -> out
  gemm_bf16_wmma<1, 1><<<gdim(NPIX, CH), 256, 0, stream>>>(
      out2, 2 * CH, proj2_w, 2 * CH, proj2_b, out, CH, NPIX, CH, 2 * CH);
}
